// GraphCastNet_20882130993565
// MI455X (gfx1250) — matrix operand order repaired and checked by reading, not compiled, not run on hardware
//
#include <hip/hip_runtime.h>
#include <hip/hip_bf16.h>

typedef __bf16 bf16_t;
typedef __attribute__((ext_vector_type(16))) __bf16 v16bf;
typedef __attribute__((ext_vector_type(8)))  float  v8f;

#define NGc   65160
#define NMc   10242
#define EMc   61440
#define EG2Mc 260640
#define EM2Gc 195480
#define Lc    256
#define VGINc 256
#define VGOUTc 64

#define ROWS_PER_BLOCK 32

struct Seg {
    const float* src;   // f32 feature rows [*, w]
    const int*   idx;   // gather index per output row (nullptr = identity)
    int w;              // real width
    int wpad;           // padded width (multiple of 32)
};

// ---------------------------------------------------------------------------
// Weight prep: f32 row-major [K][N]  ->  bf16 N-major [N][Kpad], zero-padded K
// ---------------------------------------------------------------------------
__global__ void gc_prep_wt(const float* __restrict__ W, bf16_t* __restrict__ WT,
                           int K, int Kpad, int N) {
    long i = (long)blockIdx.x * blockDim.x + threadIdx.x;
    long total = (long)N * Kpad;
    if (i >= total) return;
    int n = (int)(i / Kpad);
    int k = (int)(i - (long)n * Kpad);
    float v = (k < K) ? W[(size_t)k * N + n] : 0.0f;
    WT[i] = (bf16_t)v;
}

// ---------------------------------------------------------------------------
// WMMA fragment loaders (contiguous k order, same in A and B, wave32)
//   A 16x32: lane<16 -> row=lane,    K = k0..k0+15
//            lane>=16 -> row=lane-16, K = k0+16..k0+31
//   B 32x16 (weights stored N-major): lane<16 -> col=lane,    K = k0..k0+15
//                                     lane>=16 -> col=lane-16, K = k0+16..k0+31
// ---------------------------------------------------------------------------
__device__ __forceinline__ v16bf gc_load_A(const bf16_t* base, int stride,
                                           int k0, int half, int l16) {
    const bf16_t* row = base + l16 * stride;
    v16bf a;
#pragma unroll
    for (int i = 0; i < 16; ++i) a[i] = row[k0 + half * 16 + i];
    return a;
}

__device__ __forceinline__ v16bf gc_load_B(const bf16_t* wt, int kstride,
                                           int n, int k0, int half) {
    const bf16_t* col = wt + (size_t)n * kstride + k0 + half * 16;
    v16bf b;
#pragma unroll
    for (int i = 0; i < 16; ++i) b[i] = col[i];
    return b;
}

// ---------------------------------------------------------------------------
// Fused GraphCast MLP block:
//   x = concat(gather(seg0), gather(seg1), gather(seg2))   (bf16, LDS)
//   h = silu(x @ W1 + b1)                                  (WMMA bf16->f32)
//   y = h @ W2 + b2                                        (WMMA bf16->f32)
//   if (ln)   y = LN(y) * g + bt
//   stored = y + residual (optional)
//   if (outp) outp[row] = stored
//   if (smode) atomicAdd(agg[didx[row]], smode==1 ? y : stored)
// Block: 256 threads = 8 waves, 32 rows per block (two 16-row M-tiles per
// B-fragment load -> 2x B reuse), waves split N-tiles.
// ---------------------------------------------------------------------------
__launch_bounds__(256)
__global__ void gc_fused_mlp(Seg s0, Seg s1, Seg s2,
                             int nrows, int din_pad, int dout,
                             const bf16_t* __restrict__ WT1, const float* __restrict__ b1,
                             const bf16_t* __restrict__ WT2, const float* __restrict__ b2,
                             const float* __restrict__ ln_g, const float* __restrict__ ln_b,
                             const float* __restrict__ res, int res_stride, int res_off,
                             float* __restrict__ outp,
                             float* __restrict__ agg, const int* __restrict__ didx,
                             int smode) {
    __shared__ __align__(16) bf16_t xlds[ROWS_PER_BLOCK * 768];  // 48 KB
    __shared__ __align__(16) bf16_t hlds[ROWS_PER_BLOCK * 256];  // 16 KB
    __shared__ float  ylds[ROWS_PER_BLOCK * 256];                // 32 KB
    __shared__ float  smean[ROWS_PER_BLOCK], srstd[ROWS_PER_BLOCK];

    const int  tid     = threadIdx.x;
    const long rowbase = (long)blockIdx.x * ROWS_PER_BLOCK;

    // ---- stage gathered, padded inputs into LDS as bf16 ----
    for (int i = tid; i < ROWS_PER_BLOCK * din_pad; i += 256) {
        int  r = i / din_pad;
        int  k = i - r * din_pad;
        long grow = rowbase + r;
        float v = 0.0f;
        if (grow < nrows) {
            Seg sg; int off;
            if (k < s0.wpad)               { sg = s0; off = 0; }
            else if (k < s0.wpad + s1.wpad){ sg = s1; off = s0.wpad; }
            else                           { sg = s2; off = s0.wpad + s1.wpad; }
            int kk = k - off;
            if (kk < sg.w) {
                long srow = sg.idx ? (long)sg.idx[grow] : grow;
                v = sg.src[(size_t)srow * sg.w + kk];
            }
        }
        xlds[i] = (bf16_t)v;
    }
    __syncthreads();

    const int lane = tid & 31;
    const int wave = tid >> 5;
    const int half = lane >> 4;
    const int l16  = lane & 15;

    // ---- GEMM1: hidden (K = din_pad, N = 256), SiLU epilogue -> bf16 LDS ----
    for (int t = wave; t < 16; t += 8) {
        v8f acc0, acc1;
#pragma unroll
        for (int i = 0; i < 8; ++i) { acc0[i] = 0.0f; acc1[i] = 0.0f; }
        for (int k0 = 0; k0 < din_pad; k0 += 32) {
            v16bf b  = gc_load_B(WT1, din_pad, t * 16 + l16, k0, half);
            v16bf a0 = gc_load_A(xlds, din_pad, k0, half, l16);
            v16bf a1 = gc_load_A(xlds + 16 * din_pad, din_pad, k0, half, l16);
            acc0 = __builtin_amdgcn_wmma_f32_16x16x32_bf16(
                       false, a0, false, b, (short)0, acc0, false, false);
            acc1 = __builtin_amdgcn_wmma_f32_16x16x32_bf16(
                       false, a1, false, b, (short)0, acc1, false, false);
        }
        int n = t * 16 + l16;
        float bias = b1[n];
#pragma unroll
        for (int m = 0; m < 8; ++m) {
            int row = half * 8 + m;
            float v0 = acc0[m] + bias;
            float v1 = acc1[m] + bias;
            hlds[row * 256 + n]        = (bf16_t)(v0 / (1.0f + __expf(-v0)));
            hlds[(16 + row) * 256 + n] = (bf16_t)(v1 / (1.0f + __expf(-v1)));
        }
    }
    __syncthreads();

    // ---- GEMM2: output (K = 256, N = dout) ----
    const int ntiles = dout >> 4;
    for (int t = wave; t < ntiles; t += 8) {
        v8f acc0, acc1;
#pragma unroll
        for (int i = 0; i < 8; ++i) { acc0[i] = 0.0f; acc1[i] = 0.0f; }
        for (int k0 = 0; k0 < 256; k0 += 32) {
            v16bf b  = gc_load_B(WT2, 256, t * 16 + l16, k0, half);
            v16bf a0 = gc_load_A(hlds, 256, k0, half, l16);
            v16bf a1 = gc_load_A(hlds + 16 * 256, 256, k0, half, l16);
            acc0 = __builtin_amdgcn_wmma_f32_16x16x32_bf16(
                       false, a0, false, b, (short)0, acc0, false, false);
            acc1 = __builtin_amdgcn_wmma_f32_16x16x32_bf16(
                       false, a1, false, b, (short)0, acc1, false, false);
        }
        int n = t * 16 + l16;
        float bias = b2[n];
#pragma unroll
        for (int m = 0; m < 8; ++m) {
            int row = half * 8 + m;
            ylds[row * dout + n]        = acc0[m] + bias;
            ylds[(16 + row) * dout + n] = acc1[m] + bias;
        }
    }
    __syncthreads();

    // ---- LayerNorm over dout per row ----
    if (ln_g != nullptr) {
        if (tid < ROWS_PER_BLOCK) {
            float s = 0.0f, ss = 0.0f;
            for (int c = 0; c < dout; ++c) {
                float v = ylds[tid * dout + c];
                s += v; ss += v * v;
            }
            float mu  = s / dout;
            float var = ss / dout - mu * mu;
            smean[tid] = mu;
            srstd[tid] = rsqrtf(var + 1e-5f);
        }
        __syncthreads();
        for (int i = tid; i < ROWS_PER_BLOCK * dout; i += 256) {
            int r = i / dout, c = i - r * dout;
            ylds[i] = (ylds[i] - smean[r]) * srstd[r] * ln_g[c] + ln_b[c];
        }
        __syncthreads();
    }

    // ---- residual / store / atomic scatter-add ----
    for (int i = tid; i < ROWS_PER_BLOCK * dout; i += 256) {
        int  r = i / dout, c = i - r * dout;
        long grow = rowbase + r;
        if (grow >= nrows) continue;
        float y = ylds[i];
        float stored = y;
        if (res) stored += res[(size_t)grow * res_stride + res_off + c];
        if (outp) outp[(size_t)grow * dout + c] = stored;
        if (smode) {
            float v = (smode == 1) ? y : stored;
            atomicAdd(&agg[(size_t)didx[grow] * dout + c], v);
        }
    }
}

// ---------------------------------------------------------------------------
// Host orchestration
// ---------------------------------------------------------------------------
struct MlpP { const float *b1, *b2, *bt, *g, *w1, *w2; };

extern "C" void kernel_launch(void* const* d_in, const int* in_sizes, int n_in,
                              void* d_out, int out_size, void* d_ws, size_t ws_size,
                              hipStream_t stream) {
    (void)in_sizes; (void)n_in; (void)out_size; (void)ws_size;

    const float* grid_feats = (const float*)d_in[0];
    const float* mesh_feats = (const float*)d_in[1];
    const float* mesh_edge  = (const float*)d_in[2];
    const float* g2m_edge   = (const float*)d_in[3];
    const float* m2g_edge   = (const float*)d_in[4];
    const int* g2m_src = (const int*)d_in[5];
    const int* g2m_dst = (const int*)d_in[6];
    const int* m2m_src = (const int*)d_in[7];
    const int* m2m_dst = (const int*)d_in[8];
    const int* m2g_src = (const int*)d_in[9];
    const int* m2g_dst = (const int*)d_in[10];

    // params leaves, jax tree order (dict keys sorted alphabetically)
    int p = 11;
    auto take6 = [&]() -> MlpP {
        MlpP m;
        m.b1 = (const float*)d_in[p + 0]; m.b2 = (const float*)d_in[p + 1];
        m.bt = (const float*)d_in[p + 2]; m.g  = (const float*)d_in[p + 3];
        m.w1 = (const float*)d_in[p + 4]; m.w2 = (const float*)d_in[p + 5];
        p += 6; return m;
    };
    MlpP dec_edge = take6();
    MlpP dec_node = take6();
    MlpP dec_out;               // no LN: leaves b1,b2,w1,w2
    dec_out.b1 = (const float*)d_in[p + 0]; dec_out.b2 = (const float*)d_in[p + 1];
    dec_out.w1 = (const float*)d_in[p + 2]; dec_out.w2 = (const float*)d_in[p + 3];
    dec_out.bt = nullptr; dec_out.g = nullptr; p += 4;
    MlpP emb_eg2m = take6();
    MlpP emb_em   = take6();
    MlpP emb_em2g = take6();
    MlpP emb_vg   = take6();
    MlpP emb_vm   = take6();
    MlpP enc_edge = take6();
    MlpP enc_grid = take6();
    MlpP enc_mesh = take6();
    MlpP proc_e   = take6();    // stacked [4,...]
    MlpP proc_n   = take6();    // stacked [4,...]

    // workspace carve-out
    char* wsb = (char*)d_ws;
    size_t cur = 0;
    auto alloc = [&](size_t bytes) -> void* {
        void* r = wsb + cur;
        cur += (bytes + 255) & ~(size_t)255;
        return r;
    };
    float* vg    = (float*)alloc((size_t)NGc   * Lc * 4);
    float* vm    = (float*)alloc((size_t)NMc   * Lc * 4);
    float* em    = (float*)alloc((size_t)EMc   * Lc * 4);
    float* eg2m  = (float*)alloc((size_t)EG2Mc * Lc * 4);
    float* em2g  = (float*)alloc((size_t)EM2Gc * Lc * 4);
    float* agg_m = (float*)alloc((size_t)NMc   * Lc * 4);
    float* agg_g = (float*)alloc((size_t)NGc   * Lc * 4);

    auto prep = [&](const float* W, int K, int Kpad, int N) -> const bf16_t* {
        bf16_t* wt = (bf16_t*)alloc((size_t)N * Kpad * sizeof(bf16_t));
        long total = (long)N * Kpad;
        gc_prep_wt<<<(int)((total + 255) / 256), 256, 0, stream>>>(W, wt, K, Kpad, N);
        return wt;
    };

    const bf16_t* w_evg1  = prep(emb_vg.w1,   256, 256, 256);
    const bf16_t* w_evg2  = prep(emb_vg.w2,   256, 256, 256);
    const bf16_t* w_evm1  = prep(emb_vm.w1,     3,  32, 256);
    const bf16_t* w_evm2  = prep(emb_vm.w2,   256, 256, 256);
    const bf16_t* w_eem1  = prep(emb_em.w1,     4,  32, 256);
    const bf16_t* w_eem2  = prep(emb_em.w2,   256, 256, 256);
    const bf16_t* w_eg2m1 = prep(emb_eg2m.w1,   4,  32, 256);
    const bf16_t* w_eg2m2 = prep(emb_eg2m.w2, 256, 256, 256);
    const bf16_t* w_em2g1 = prep(emb_em2g.w1,   4,  32, 256);
    const bf16_t* w_em2g2 = prep(emb_em2g.w2, 256, 256, 256);
    const bf16_t* w_ence1 = prep(enc_edge.w1, 768, 768, 256);
    const bf16_t* w_ence2 = prep(enc_edge.w2, 256, 256, 256);
    const bf16_t* w_encm1 = prep(enc_mesh.w1, 512, 512, 256);
    const bf16_t* w_encm2 = prep(enc_mesh.w2, 256, 256, 256);
    const bf16_t* w_encg1 = prep(enc_grid.w1, 256, 256, 256);
    const bf16_t* w_encg2 = prep(enc_grid.w2, 256, 256, 256);
    const bf16_t* w_pe1[4], *w_pe2[4], *w_pn1[4], *w_pn2[4];
    for (int s = 0; s < 4; ++s) {
        w_pe1[s] = prep(proc_e.w1 + (size_t)s * 768 * 256, 768, 768, 256);
        w_pe2[s] = prep(proc_e.w2 + (size_t)s * 256 * 256, 256, 256, 256);
        w_pn1[s] = prep(proc_n.w1 + (size_t)s * 512 * 256, 512, 512, 256);
        w_pn2[s] = prep(proc_n.w2 + (size_t)s * 256 * 256, 256, 256, 256);
    }
    const bf16_t* w_dece1 = prep(dec_edge.w1, 768, 768, 256);
    const bf16_t* w_dece2 = prep(dec_edge.w2, 256, 256, 256);
    const bf16_t* w_decn1 = prep(dec_node.w1, 512, 512, 256);
    const bf16_t* w_decn2 = prep(dec_node.w2, 256, 256, 256);
    const bf16_t* w_deco1 = prep(dec_out.w1,  256, 256, 256);
    const bf16_t* w_deco2 = prep(dec_out.w2,  256, 256,  64);

    const Seg Z = {nullptr, nullptr, 0, 0};
    auto run = [&](Seg a, Seg b, Seg c, int nrows, int din_pad, int dout,
                   const bf16_t* wt1, const float* bb1, const bf16_t* wt2, const float* bb2,
                   const float* g, const float* bt,
                   const float* res, int rs, int ro, float* outp,
                   float* agg, const int* didx, int smode) {
        int blocks = (nrows + ROWS_PER_BLOCK - 1) / ROWS_PER_BLOCK;
        gc_fused_mlp<<<blocks, 256, 0, stream>>>(a, b, c, nrows, din_pad, dout,
                                                 wt1, bb1, wt2, bb2, g, bt,
                                                 res, rs, ro, outp, agg, didx, smode);
    };

    // ---- encoder embeddings ----
    run({grid_feats, nullptr, 256, 256}, Z, Z, NGc, 256, 256,
        w_evg1, emb_vg.b1, w_evg2, emb_vg.b2, emb_vg.g, emb_vg.bt,
        nullptr, 0, 0, vg, nullptr, nullptr, 0);
    run({mesh_feats, nullptr, 3, 32}, Z, Z, NMc, 32, 256,
        w_evm1, emb_vm.b1, w_evm2, emb_vm.b2, emb_vm.g, emb_vm.bt,
        nullptr, 0, 0, vm, nullptr, nullptr, 0);
    run({mesh_edge, nullptr, 4, 32}, Z, Z, EMc, 32, 256,
        w_eem1, emb_em.b1, w_eem2, emb_em.b2, emb_em.g, emb_em.bt,
        nullptr, 0, 0, em, nullptr, nullptr, 0);
    run({g2m_edge, nullptr, 4, 32}, Z, Z, EG2Mc, 32, 256,
        w_eg2m1, emb_eg2m.b1, w_eg2m2, emb_eg2m.b2, emb_eg2m.g, emb_eg2m.bt,
        nullptr, 0, 0, eg2m, nullptr, nullptr, 0);
    run({m2g_edge, nullptr, 4, 32}, Z, Z, EM2Gc, 32, 256,
        w_em2g1, emb_em2g.b1, w_em2g2, emb_em2g.b2, emb_em2g.g, emb_em2g.bt,
        nullptr, 0, 0, em2g, nullptr, nullptr, 0);

    // ---- encoder: grid->mesh message passing ----
    hipMemsetAsync(agg_m, 0, (size_t)NMc * Lc * 4, stream);
    run({eg2m, nullptr, 256, 256}, {vg, g2m_src, 256, 256}, {vm, g2m_dst, 256, 256},
        EG2Mc, 768, 256, w_ence1, enc_edge.b1, w_ence2, enc_edge.b2,
        enc_edge.g, enc_edge.bt,
        eg2m, 256, 0, eg2m, agg_m, g2m_dst, /*scatter pre-residual e*/ 1);
    run({vm, nullptr, 256, 256}, {agg_m, nullptr, 256, 256}, Z,
        NMc, 512, 256, w_encm1, enc_mesh.b1, w_encm2, enc_mesh.b2,
        enc_mesh.g, enc_mesh.bt, vm, 256, 0, vm, nullptr, nullptr, 0);
    run({vg, nullptr, 256, 256}, Z, Z,
        NGc, 256, 256, w_encg1, enc_grid.b1, w_encg2, enc_grid.b2,
        enc_grid.g, enc_grid.bt, vg, 256, 0, vg, nullptr, nullptr, 0);

    // ---- processor: 4 mesh GNN steps ----
    for (int s = 0; s < 4; ++s) {
        hipMemsetAsync(agg_m, 0, (size_t)NMc * Lc * 4, stream);
        run({em, nullptr, 256, 256}, {vm, m2m_src, 256, 256}, {vm, m2m_dst, 256, 256},
            EMc, 768, 256, w_pe1[s], proc_e.b1 + s * 256, w_pe2[s], proc_e.b2 + s * 256,
            proc_e.g + s * 256, proc_e.bt + s * 256,
            em, 256, 0, em, agg_m, m2m_dst, /*scatter updated em*/ 2);
        run({vm, nullptr, 256, 256}, {agg_m, nullptr, 256, 256}, Z,
            NMc, 512, 256, w_pn1[s], proc_n.b1 + s * 256, w_pn2[s], proc_n.b2 + s * 256,
            proc_n.g + s * 256, proc_n.bt + s * 256,
            vm, 256, 0, vm, nullptr, nullptr, 0);
    }

    // ---- decoder: mesh->grid message passing + output head ----
    hipMemsetAsync(agg_g, 0, (size_t)NGc * Lc * 4, stream);
    run({em2g, nullptr, 256, 256}, {vm, m2g_src, 256, 256}, {vg, m2g_dst, 256, 256},
        EM2Gc, 768, 256, w_dece1, dec_edge.b1, w_dece2, dec_edge.b2,
        dec_edge.g, dec_edge.bt,
        em2g, 256, 0, em2g, agg_g, m2g_dst, /*scatter updated em2g*/ 2);
    run({vg, nullptr, 256, 256}, {agg_g, nullptr, 256, 256}, Z,
        NGc, 512, 256, w_decn1, dec_node.b1, w_decn2, dec_node.b2,
        dec_node.g, dec_node.bt, vg, 256, 0, vg, nullptr, nullptr, 0);
    run({vg, nullptr, 256, 256}, Z, Z,
        NGc, 256, VGOUTc, w_deco1, dec_out.b1, w_deco2, dec_out.b2,
        nullptr, nullptr,
        grid_feats, VGINc, VGOUTc, (float*)d_out, nullptr, nullptr, 0);
}
